// LinearAttention_38336878084670
// MI455X (gfx1250) — matrix operand order repaired and checked by reading, not compiled, not run on hardware
//
#include <hip/hip_runtime.h>

// ---------------------------------------------------------------------------
// LinearAttention for MI455X (gfx1250): bf16-input / f32-accumulate WMMA path.
// Fragment-ready LDS layouts (swizzle on store side -> ds_load_b128 operand
// reads), software-pipelined global loads, branchless bias epilogue.
// ---------------------------------------------------------------------------

typedef __bf16 bf16_t;
typedef bf16_t v16bf __attribute__((ext_vector_type(16)));
typedef float  v8f   __attribute__((ext_vector_type(8)));

#define BATCH 8
#define CIN   256
#define NPIX  (128 * 128)   // 16384
#define HID   128
#define HEADS 4
#define DIMH  32
#define OC3   384           // 3 * HID

// Pack two f32 into one dword of 2 x bf16 (consecutive-K pair).
__device__ __forceinline__ uint32_t pk_bf16(float lo, float hi) {
#if __has_builtin(__builtin_amdgcn_cvt_pk_bf16_f32)
    auto r = __builtin_amdgcn_cvt_pk_bf16_f32(lo, hi);
    union { decltype(r) h; uint32_t u; } p;
    p.h = r;
    return p.u;
#else
    union { bf16_t h[2]; uint32_t u; } p;
    p.h[0] = (bf16_t)lo;
    p.h[1] = (bf16_t)hi;
    return p.u;
#endif
}

// Read a lane's 16-bf16 fragment (8 dwords, 32B, 16B-aligned) from LDS.
__device__ __forceinline__ v16bf read_frag(const uint32_t* s) {
    return *reinterpret_cast<const v16bf*>(s);
}

__device__ __forceinline__ v8f wmma_bf16(v16bf a, v16bf b, v8f c) {
    return __builtin_amdgcn_wmma_f32_16x16x32_bf16(
        /*neg_a=*/false, a, /*neg_b=*/false, b,
        /*c_mod=*/(short)0, c, /*reuse_a=*/false, /*reuse_b=*/false);
}

// ---------------------------------------------------------------------------
// Fragment-layout index math (CDNA5 ISA, 16-bit operands, wave32).
// Element = K-pair index p (0..15 for K=32) placed into (lane, dword v):
//  A 16x32 (MxK), row m:   lane = (m&15) + 16*((p>>2)&1), v = (p&3) + 4*(p>>3)
//  B 32x16 (KxN), col n:   lane = (n&15) + 16*((p>>2)&1), v = (p&3) + 4*(p>>3)
// ---------------------------------------------------------------------------
__device__ __forceinline__ int a_lane(int m, int p) { return (m & 15) + (((p >> 2) & 1) << 4); }
__device__ __forceinline__ int a_dw  (int p)        { return (p & 3) + ((p >> 3) << 2); }
__device__ __forceinline__ int b_lane(int n, int p) { return (n & 15) + (((p >> 2) & 1) << 4); }
__device__ __forceinline__ int b_dw  (int p)        { return (p & 3) + ((p >> 3) << 2); }

// ---------------------------------------------------------------------------
// Generic tiled GEMM: C[z] = A(MxK, fp32) * B[z](KxN, fp32) (+ bias)
// Block = 128 threads (4 waves). Tile BM=64, BN=64, BK=32.
// Software-pipelined: next K-tile global loads issue under current WMMAs.
// Requires M%64==0, K%32==0, N%64==0.
// ---------------------------------------------------------------------------
#define BM 64
#define BN 64
#define BK 32

template <bool HAS_BIAS>
__global__ __launch_bounds__(128)
void gemm_bf16_kernel(const float* __restrict__ A, const float* __restrict__ Bg,
                      float* __restrict__ Cg, const float* __restrict__ bias,
                      int M, int K, int N, long strideB, long strideC) {
    __shared__ __align__(16) uint32_t AsF[4][32][8];   // 4 M-subtiles, frag-ready
    __shared__ __align__(16) uint32_t BsF[4][32][8];   // 4 N-subtiles, frag-ready

    const int tid  = threadIdx.x;
    const int lane = tid & 31;
    const int wv   = tid >> 5;                 // 0..3 -> M subtile
    const int tm   = blockIdx.y * BM;
    const int tn   = blockIdx.x * BN;
    const float* Bp = Bg + (long)blockIdx.z * strideB;
    float*       Cp = Cg + (long)blockIdx.z * strideC;

    // K-invariant per-thread LDS destinations and global source cursors.
    uint32_t* aDst[8];
    uint32_t* bDst[8];
    const float* aSrc[8];
    const float* bSrc[8];
#pragma unroll
    for (int i = 0; i < 8; ++i) {
        const int idx = tid + i * 128;
        const int m = idx >> 4, p = idx & 15;  // A: row m, K-pair p
        aDst[i] = &AsF[m >> 4][a_lane(m, p)][a_dw(p)];
        aSrc[i] = &A[(long)(tm + m) * K + 2 * p];
        const int n = idx & 63, q = idx >> 6;  // B: col n, K-pair q
        bDst[i] = &BsF[n >> 4][b_lane(n, q)][b_dw(q)];
        bSrc[i] = &Bp[(long)(2 * q) * N + tn + n];
    }

    v8f acc[4] = {};
    float2 aReg[8];
    float  bLo[8], bHi[8];

    // Prologue: fetch K-tile 0 into registers.
#pragma unroll
    for (int i = 0; i < 8; ++i) {
        aReg[i] = *reinterpret_cast<const float2*>(aSrc[i]);
        bLo[i]  = bSrc[i][0];
        bHi[i]  = bSrc[i][(long)N];
    }

    const int KT = K / BK;
    for (int kt = 0; kt < KT; ++kt) {
        // Convert + scatter current tile into fragment-ready LDS.
#pragma unroll
        for (int i = 0; i < 8; ++i) {
            *aDst[i] = pk_bf16(aReg[i].x, aReg[i].y);
            *bDst[i] = pk_bf16(bLo[i], bHi[i]);
        }
        __syncthreads();

        // Issue next tile's global loads; they complete under the WMMAs.
        if (kt + 1 < KT) {
#pragma unroll
            for (int i = 0; i < 8; ++i) {
                aSrc[i] += BK;
                bSrc[i] += (long)BK * N;
                aReg[i] = *reinterpret_cast<const float2*>(aSrc[i]);
                bLo[i]  = bSrc[i][0];
                bHi[i]  = bSrc[i][(long)N];
            }
        }

        const v16bf af = read_frag(AsF[wv][lane]);
#pragma unroll
        for (int nt = 0; nt < 4; ++nt) {
            acc[nt] = wmma_bf16(af, read_frag(BsF[nt][lane]), acc[nt]);
        }
        __syncthreads();
    }

    const int chi = lane >> 4, col0 = lane & 15;
    float brow[8];
    if (HAS_BIAS) {
#pragma unroll
        for (int r = 0; r < 8; ++r) brow[r] = bias[tm + wv * 16 + r + 8 * chi];
    }
#pragma unroll
    for (int nt = 0; nt < 4; ++nt) {
#pragma unroll
        for (int r = 0; r < 8; ++r) {
            const int row = tm + wv * 16 + r + 8 * chi;
            const int col = tn + nt * 16 + col0;
            float vv = acc[nt][r];
            if (HAS_BIAS) vv += brow[r];
            Cp[(long)row * N + col] = vv;
        }
    }
}

// ---------------------------------------------------------------------------
// Softmax over spatial n for each k-row (rows 128..255 of qkv), in place, f32.
// One 256-thread block per row; B*HID = 1024 rows. float4 streaming.
// ---------------------------------------------------------------------------
__global__ __launch_bounds__(256)
void softmax_rows_kernel(float* __restrict__ qkv) {
    const int b = blockIdx.x >> 7;             // /128
    const int r = blockIdx.x & 127;
    float* row = qkv + ((long)b * OC3 + HID + r) * (long)NPIX;
    float4* row4 = reinterpret_cast<float4*>(row);
    const int NV = NPIX / 4;                   // 4096 float4

    __shared__ float red[256];
    const int tid = threadIdx.x;

    float m = -3.402823466e38f;
    for (int i = tid; i < NV; i += 256) {
        const float4 v = row4[i];
        m = fmaxf(m, fmaxf(fmaxf(v.x, v.y), fmaxf(v.z, v.w)));
    }
    red[tid] = m;
    __syncthreads();
    for (int s = 128; s > 0; s >>= 1) {
        if (tid < s) red[tid] = fmaxf(red[tid], red[tid + s]);
        __syncthreads();
    }
    m = red[0];
    __syncthreads();

    float sum = 0.f;
    for (int i = tid; i < NV; i += 256) {
        const float4 v = row4[i];
        sum += __expf(v.x - m) + __expf(v.y - m) + __expf(v.z - m) + __expf(v.w - m);
    }
    red[tid] = sum;
    __syncthreads();
    for (int s = 128; s > 0; s >>= 1) {
        if (tid < s) red[tid] += red[tid + s];
        __syncthreads();
    }
    const float inv = 1.f / red[0];

    for (int i = tid; i < NV; i += 256) {
        float4 v = row4[i];
        v.x = __expf(v.x - m) * inv;
        v.y = __expf(v.y - m) * inv;
        v.z = __expf(v.z - m) * inv;
        v.w = __expf(v.w - m) * inv;
        row4[i] = v;
    }
}

// ---------------------------------------------------------------------------
// context[b,h,d,e] = sum_n softmax_k[d,n] * v[e,n]   (32x32, K = 16384)
// One block per (b,h): 8 waves, each reduces N/8 columns with 2x2 WMMA tiles,
// then cross-wave reduction through LDS. The v-transpose is folded into the
// fragment-store index math (B = n x e built directly from [e][n] source).
// ---------------------------------------------------------------------------
struct CtxStage {
    uint32_t kF[8][2][32][8];   // per wave, 2 M-subtiles (d), frag-ready
    uint32_t vF[8][2][32][8];   // per wave, 2 N-subtiles (e), frag-ready
};

__global__ __launch_bounds__(256)
void context_kernel(const float* __restrict__ qkv, float* __restrict__ ctx) {
    const int bh = blockIdx.x;                 // 0..31
    const int b = bh >> 2, h = bh & 3;
    const float* kg = qkv + ((long)b * OC3 + HID + h * DIMH) * (long)NPIX;
    const float* vg = qkv + ((long)b * OC3 + 2 * HID + h * DIMH) * (long)NPIX;

    __shared__ __align__(16) union { CtxStage st; float red[8 * 1024]; } sm;

    const int tid = threadIdx.x, lane = tid & 31, wv = tid >> 5;

    v8f acc[2][2] = {};
    const int lhalf = lane >> 4;               // row parity handled per lane
    const int p     = lane & 15;               // K-pair handled per lane

    // Step-invariant LDS destinations (hoisted out of the reduction loop).
    uint32_t* kD[16];
    uint32_t* vD[16];
#pragma unroll
    for (int j = 0; j < 16; ++j) {
        const int row = 2 * j + lhalf;         // d for k, e for v (0..31)
        kD[j] = &sm.st.kF[wv][row >> 4][a_lane(row, p)][a_dw(p)];
        vD[j] = &sm.st.vF[wv][row >> 4][b_lane(row, p)][b_dw(p)];
    }
    const float* kBase = kg + (long)lhalf * NPIX + 2 * p + (long)wv * (NPIX / 8);
    const float* vBase = vg + (long)lhalf * NPIX + 2 * p + (long)wv * (NPIX / 8);

    for (int step = 0; step < (NPIX / 8) / 32; ++step) {
        const long n0 = (long)step * 32;
        // Stage k (A: d x n) and v (B: n x e) panels, 32x32 each.
#pragma unroll
        for (int j = 0; j < 16; ++j) {
            const float2 ks = *reinterpret_cast<const float2*>(
                &kBase[(long)(2 * j) * NPIX + n0]);
            const float2 vs = *reinterpret_cast<const float2*>(
                &vBase[(long)(2 * j) * NPIX + n0]);
            *kD[j] = pk_bf16(ks.x, ks.y);
            *vD[j] = pk_bf16(vs.x, vs.y);
        }
#pragma unroll
        for (int mt = 0; mt < 2; ++mt) {
            const v16bf af = read_frag(sm.st.kF[wv][mt][lane]);
#pragma unroll
            for (int nt = 0; nt < 2; ++nt) {
                acc[mt][nt] = wmma_bf16(af, read_frag(sm.st.vF[wv][nt][lane]),
                                        acc[mt][nt]);
            }
        }
    }

    __syncthreads();                           // stage LDS -> reduction LDS
    const int chi = lane >> 4, col0 = lane & 15;
#pragma unroll
    for (int mt = 0; mt < 2; ++mt)
#pragma unroll
        for (int nt = 0; nt < 2; ++nt)
#pragma unroll
            for (int r = 0; r < 8; ++r) {
                const int d = mt * 16 + r + 8 * chi;
                const int e = nt * 16 + col0;
                sm.red[wv * 1024 + d * 32 + e] = acc[mt][nt][r];
            }
    __syncthreads();

    for (int i = tid; i < 1024; i += 256) {
        float s = 0.f;
#pragma unroll
        for (int w = 0; w < 8; ++w) s += sm.red[w * 1024 + i];
        ctx[(long)bh * 1024 + i] = s;
    }
}

// ---------------------------------------------------------------------------
// att[b, h*32+e, n] = sum_d context[b,h,d,e] * q[b, h*32+d, n]
// A = context^T (built frag-ready from [d][e] source), B = q panel (32 x 128).
// K = 32 -> exactly one WMMA step. Grid: (NPIX/128, 32). 256 threads, 8 waves.
// ---------------------------------------------------------------------------
__global__ __launch_bounds__(256)
void apply_attn_kernel(const float* __restrict__ qkv,
                       const float* __restrict__ ctx,
                       float* __restrict__ att) {
    const int bh = blockIdx.y;
    const int b = bh >> 2, h = bh & 3;
    const long n_base = (long)blockIdx.x * 128;
    const float* qg = qkv + ((long)b * OC3 + h * DIMH) * (long)NPIX;
    const float* cx = ctx + (long)bh * 1024;

    __shared__ __align__(16) uint32_t ctxF[2][32][8];  // A: M=e subtiles, K=d
    __shared__ __align__(16) uint32_t qF[8][32][8];    // B: 8 n-subtiles, K=d

    const int tid = threadIdx.x, lane = tid & 31, wv = tid >> 5;

    // context^T fragments: element (e, d-pair p): src ctx[d*32 + e]
#pragma unroll
    for (int i = 0; i < 2; ++i) {
        const int idx = tid + i * 256;         // 0..511
        const int e = idx & 31;
        const int p = idx >> 5;                // d-pair
        const float lo = cx[(2 * p) * 32 + e];
        const float hi = cx[(2 * p + 1) * 32 + e];
        ctxF[e >> 4][a_lane(e, p)][a_dw(p)] = pk_bf16(lo, hi);
    }
    // q fragments: element (n, d-pair p): src qg[d*NPIX + n_base + n]
#pragma unroll
    for (int i = 0; i < 8; ++i) {
        const int idx = tid + i * 256;         // 0..2047
        const int n = idx & 127;
        const int p = idx >> 7;
        const float lo = qg[(long)(2 * p) * NPIX + n_base + n];
        const float hi = qg[(long)(2 * p + 1) * NPIX + n_base + n];
        qF[n >> 4][b_lane(n, p)][b_dw(p)] = pk_bf16(lo, hi);
    }
    __syncthreads();

    v8f acc[2] = {};
    const v16bf bfr = read_frag(qF[wv][lane]);
#pragma unroll
    for (int mt = 0; mt < 2; ++mt) {
        acc[mt] = wmma_bf16(read_frag(ctxF[mt][lane]), bfr, acc[mt]);
    }

    float* ag = att + ((long)b * HID + h * DIMH) * (long)NPIX;
    const int chi = lane >> 4, col0 = lane & 15;
#pragma unroll
    for (int mt = 0; mt < 2; ++mt)
#pragma unroll
        for (int r = 0; r < 8; ++r) {
            const int e = mt * 16 + r + 8 * chi;
            ag[(long)e * NPIX + n_base + wv * 16 + col0] = acc[mt][r];
        }
}

// ---------------------------------------------------------------------------
// Launcher
// ---------------------------------------------------------------------------
extern "C" void kernel_launch(void* const* d_in, const int* in_sizes, int n_in,
                              void* d_out, int out_size, void* d_ws, size_t ws_size,
                              hipStream_t stream) {
    const float* x     = (const float*)d_in[0];  // [8,256,128,128]
    const float* w_qkv = (const float*)d_in[1];  // [384,256]
    const float* w_out = (const float*)d_in[2];  // [256,128]
    const float* b_out = (const float*)d_in[3];  // [256]
    float* out = (float*)d_out;                  // [8,256,128,128]

    float* qkv = (float*)d_ws;                               // 8*384*16384 f32
    float* ctx = qkv + (long)BATCH * OC3 * NPIX;             // 8*4*32*32 f32
    float* att = ctx + (long)BATCH * HEADS * DIMH * DIMH;    // 8*128*16384 f32

    // 1) qkv = w_qkv @ x          (M=384, K=256, N=16384, z=8)
    gemm_bf16_kernel<false><<<dim3(NPIX / BN, OC3 / BM, BATCH), 128, 0, stream>>>(
        w_qkv, x, qkv, nullptr, OC3, CIN, NPIX,
        (long)CIN * NPIX, (long)OC3 * NPIX);

    // 2) softmax over n on k rows (in place)
    softmax_rows_kernel<<<dim3(BATCH * HID), 256, 0, stream>>>(qkv);

    // 3) context = softmax(k) @ v^T   (32x32 per (b,h))
    context_kernel<<<dim3(BATCH * HEADS), 256, 0, stream>>>(qkv, ctx);

    // 4) att = context^T @ q
    apply_attn_kernel<<<dim3(NPIX / 128, BATCH * HEADS), 256, 0, stream>>>(
        qkv, ctx, att);

    // 5) out = w_out @ att + b_out   (M=256, K=128, N=16384, z=8)
    gemm_bf16_kernel<true><<<dim3(NPIX / BN, CIN / BM, BATCH), 128, 0, stream>>>(
        w_out, att, out, b_out, CIN, HID, NPIX,
        (long)HID * NPIX, (long)CIN * NPIX);
}
